// CrossModalAttention_76699525972289
// MI455X (gfx1250) — matrix operand order repaired and checked by reading, not compiled, not run on hardware
//
#include <hip/hip_runtime.h>

typedef __attribute__((ext_vector_type(16))) __bf16 v16bf;
typedef __attribute__((ext_vector_type(8)))  float  v8f;

constexpr int cM = 3, cB = 8, cN = 512, cC = 512, cH = 8, cHD = 64;
constexpr int BN = cB * cN;  // 4096 rows in the flattened projection GEMMs

#define DEVFN __device__ __forceinline__

// float -> bf16, round-to-nearest-even (no reliance on __bf16 arithmetic)
DEVFN __bf16 f2bf(float f) {
  union { float f; unsigned u; } a; a.f = f;
  unsigned r = a.u + 0x7FFFu + ((a.u >> 16) & 1u);
  union { unsigned short s; __bf16 b; } o;
  o.s = (unsigned short)(r >> 16);
  return o.b;
}

// A-fragment, 16x32 bf16 (row-major source, stride `ld` elements).
// Lane l: row = l&15; K in {ko..ko+7} U {16+ko..16+ko+7}, ko = (l>>4)*8.
DEVFN v16bf load_a_frag(const __bf16* tile, int ld, int lane) {
  const int r  = lane & 15;
  const int ko = (lane >> 4) * 8;
  const __bf16* p = tile + (size_t)r * ld + ko;
  union { uint4 u[2]; v16bf v; } t;
  t.u[0] = *(const uint4*)(p);        // K = ko..ko+7
  t.u[1] = *(const uint4*)(p + 16);   // K = 16+ko..16+ko+7
  return t.v;
}

// B-fragment, 32x16 bf16 from a COLUMN-MAJOR source (column stride ldT elems).
// Lane l: col = l&15; K in {ko..ko+7} U {16+ko..16+ko+7}, ko = (l>>4)*8.
DEVFN v16bf load_b_frag(const __bf16* tileT, int ldT, int lane) {
  const int c  = lane & 15;
  const int ko = (lane >> 4) * 8;
  const __bf16* p = tileT + (size_t)c * ldT + ko;
  union { uint4 u[2]; v16bf v; } t;
  t.u[0] = *(const uint4*)(p);
  t.u[1] = *(const uint4*)(p + 16);
  return t.v;
}

DEVFN v8f wmma_bf16(v16bf a, v16bf b, v8f c) {
  return __builtin_amdgcn_wmma_f32_16x16x32_bf16(
      /*neg_a=*/false, a, /*neg_b=*/false, b,
      /*c_mod=*/(short)0, c, /*reuse_a=*/false, /*reuse_b=*/false);
}

// ---------------- prep: f32 -> bf16 convert / transpose ----------------

__global__ void cvt_bf16_kernel(const float* __restrict__ src,
                                __bf16* __restrict__ dst, int n) {
  int i = blockIdx.x * blockDim.x + threadIdx.x;
  if (i < n) dst[i] = f2bf(src[i]);
}

// nmat CxC matrices: dst[mat][n][k] = bf16(src[mat][k][n])  (column-major copy)
__global__ void transpose_bf16_kernel(const float* __restrict__ src,
                                      __bf16* __restrict__ dst, int nmat) {
  int i = blockIdx.x * blockDim.x + threadIdx.x;
  int total = nmat * cC * cC;
  if (i >= total) return;
  int mat = i / (cC * cC);
  int rem = i - mat * (cC * cC);
  int k = rem / cC, n = rem - k * cC;
  dst[(size_t)mat * cC * cC + (size_t)n * cC + k] = f2bf(src[i]);
}

// ---------------- QKV projection: Y = X W + b ----------------
// out layout: transposedOut==0 -> [M,B,H,N,hd] (Q,K); ==1 -> [M,B,H,hd,N] (V)

__global__ __launch_bounds__(32) void qkv_proj_kernel(
    const __bf16* __restrict__ xb, const __bf16* __restrict__ WT,
    const float* __restrict__ bias, __bf16* __restrict__ outp,
    int transposedOut) {
  const int lane = threadIdx.x;
  const int rt = blockIdx.x, ct = blockIdx.y, m = blockIdx.z;
  const __bf16* Arow = xb + ((size_t)m * BN + (size_t)rt * 16) * cC;
  const __bf16* Bcol = WT + (size_t)m * cC * cC + (size_t)(ct * 16) * cC;
  v8f acc = {};
#pragma unroll 4
  for (int kt = 0; kt < cC / 32; ++kt) {
    v16bf a = load_a_frag(Arow + kt * 32, cC, lane);
    v16bf b = load_b_frag(Bcol + kt * 32, cC, lane);
    acc = wmma_bf16(a, b, acc);
  }
  const int col = ct * 16 + (lane & 15);
  const float bv = bias[m * cC + col];
  const int rbase = (lane >> 4) * 8;
  const int h = col >> 6, d = col & (cHD - 1);
#pragma unroll
  for (int p = 0; p < 8; ++p) {
    int row = rt * 16 + rbase + p;       // row in [0, B*N)
    int b_ = row >> 9, n = row & (cN - 1);
    float v = acc[p] + bv;
    size_t idx = transposedOut
        ? ((((size_t)m * cB + b_) * cH + h) * cHD + d) * cN + n
        : ((((size_t)m * cB + b_) * cH + h) * cN + n) * cHD + d;
    outp[idx] = f2bf(v);
  }
}

// ---------------- all-pair cross attention ----------------
// block = 1 wave; grid = (N/16 query tiles, B*H).
// fused[b,n,c] = (1/M) * sum_{i,j} mw[i,j] * softmax(Qi Kj^T /8) Vj

__global__ __launch_bounds__(32) void cross_attn_kernel(
    const __bf16* __restrict__ qb, const __bf16* __restrict__ kb,
    const __bf16* __restrict__ vtb, const float* __restrict__ mwp,
    __bf16* __restrict__ fusedb) {
  __shared__ float  sS[16][cN];   // 32 KB: f32 score/exps tile
  __shared__ __bf16 sP[16][cN];   // 16 KB: bf16 probs (A-matrix for P*V)
  const int lane = threadIdx.x;
  const int qt = blockIdx.x;
  const int b  = blockIdx.y >> 3;     // / H
  const int h  = blockIdx.y & (cH - 1);
  const int rbase = (lane >> 4) * 8;  // C/D row base; also A K-offset
  const int colc  = lane & 15;
  const float scale = 0.125f;         // hd^-0.5

  v8f accO[4] = {};

  for (int i = 0; i < cM; ++i) {
    const __bf16* qtile =
        qb + ((((size_t)i * cB + b) * cH + h) * cN + (size_t)qt * 16) * cHD;
    const v16bf aq0 = load_a_frag(qtile, cHD, lane);        // K = 0..31 of hd
    const v16bf aq1 = load_a_frag(qtile + 32, cHD, lane);   // K = 32..63
    for (int j = 0; j < cM; ++j) {
      // ---- scores S = Q K^T (16 x 512) ----
      const __bf16* kmat =
          kb + (((size_t)j * cB + b) * cH + h) * (size_t)cN * cHD;
      for (int nt = 0; nt < cN / 16; ++nt) {
        const __bf16* ktile = kmat + (size_t)(nt * 16) * cHD;
        v8f s = {};
        s = wmma_bf16(aq0, load_b_frag(ktile, cHD, lane), s);
        s = wmma_bf16(aq1, load_b_frag(ktile + 32, cHD, lane), s);
#pragma unroll
        for (int p = 0; p < 8; ++p)
          sS[rbase + p][nt * 16 + colc] = s[p] * scale;
      }
      __syncthreads();

      // ---- softmax over 512 keys; 2 lanes per row; fold mw[i,j]/M ----
      const float w = mwp[i * cM + j] * (1.0f / cM);
      const int r = lane & 15;
      const int c0 = (lane >> 4) * 256;
      float mx = -3.40282347e38f;
      for (int c = c0; c < c0 + 256; ++c) mx = fmaxf(mx, sS[r][c]);
      mx = fmaxf(mx, __shfl_xor(mx, 16, 32));
      float sum = 0.f;
      for (int c = c0; c < c0 + 256; ++c) {
        float e = __expf(sS[r][c] - mx);
        sS[r][c] = e;
        sum += e;
      }
      sum += __shfl_xor(sum, 16, 32);
      const float inv = w / sum;
      for (int c = c0; c < c0 + 256; ++c) sP[r][c] = f2bf(sS[r][c] * inv);
      __syncthreads();

      // ---- O += P (16x512) * Vj (512x64) ----
      const __bf16* vmat =
          vtb + (((size_t)j * cB + b) * cH + h) * (size_t)cHD * cN;
      for (int dt = 0; dt < 4; ++dt) {
        const __bf16* vcol = vmat + (size_t)(dt * 16) * cN;
        for (int kt = 0; kt < cN / 32; ++kt) {
          const __bf16* pr = &sP[lane & 15][kt * 32 + rbase];  // LDS A-frag
          union { uint4 u[2]; v16bf v; } ua;
          ua.u[0] = *(const uint4*)pr;
          ua.u[1] = *(const uint4*)(pr + 16);
          v16bf bv = load_b_frag(vcol + kt * 32, cN, lane);
          accO[dt] = wmma_bf16(ua.v, bv, accO[dt]);
        }
      }
      __syncthreads();
    }
  }

  // store fused tile as bf16 [B,N,C] (c = h*hd + d)
#pragma unroll
  for (int dt = 0; dt < 4; ++dt) {
#pragma unroll
    for (int p = 0; p < 8; ++p) {
      int n = qt * 16 + rbase + p;
      int c = h * cHD + dt * 16 + colc;
      fusedb[((size_t)b * cN + n) * cC + c] = f2bf(accO[dt][p]);
    }
  }
}

// ---------------- output projection: out = fused Wp + bp (f32 out) ----------------

__global__ __launch_bounds__(32) void out_proj_kernel(
    const __bf16* __restrict__ fusedb, const __bf16* __restrict__ WpT,
    const float* __restrict__ bp, float* __restrict__ out) {
  const int lane = threadIdx.x;
  const int rt = blockIdx.x, ct = blockIdx.y;
  const __bf16* Arow = fusedb + (size_t)(rt * 16) * cC;
  const __bf16* Bcol = WpT + (size_t)(ct * 16) * cC;
  v8f acc = {};
#pragma unroll 4
  for (int kt = 0; kt < cC / 32; ++kt) {
    v16bf a = load_a_frag(Arow + kt * 32, cC, lane);
    v16bf b = load_b_frag(Bcol + kt * 32, cC, lane);
    acc = wmma_bf16(a, b, acc);
  }
  const int col = ct * 16 + (lane & 15);
  const float bias = bp[col];
  const int rbase = (lane >> 4) * 8;
#pragma unroll
  for (int p = 0; p < 8; ++p) {
    int row = rt * 16 + rbase + p;
    out[(size_t)row * cC + col] = acc[p] + bias;
  }
}

// ---------------- host launcher ----------------

extern "C" void kernel_launch(void* const* d_in, const int* in_sizes, int n_in,
                              void* d_out, int out_size, void* d_ws,
                              size_t ws_size, hipStream_t stream) {
  (void)in_sizes; (void)n_in; (void)out_size; (void)ws_size;
  const float* x  = (const float*)d_in[0];
  const float* Wq = (const float*)d_in[1];
  const float* bq = (const float*)d_in[2];
  const float* Wk = (const float*)d_in[3];
  const float* bk = (const float*)d_in[4];
  const float* Wv = (const float*)d_in[5];
  const float* bv = (const float*)d_in[6];
  const float* mw = (const float*)d_in[7];
  const float* Wp = (const float*)d_in[8];
  const float* bp = (const float*)d_in[9];
  float* out = (float*)d_out;

  // carve workspace (bf16 intermediates; everything written before read)
  char* ws = (char*)d_ws;
  size_t off = 0;
  auto carve = [&](size_t bytes) -> void* {
    void* p = ws + off;
    off += (bytes + 255) & ~(size_t)255;
    return p;
  };
  const size_t nX  = (size_t)cM * cB * cN * cC;   // 6,291,456
  const size_t nW3 = (size_t)cM * cC * cC;        //   786,432
  const size_t nW1 = (size_t)cC * cC;             //   262,144
  const size_t nF  = (size_t)cB * cN * cC;        // 2,097,152

  __bf16* xb   = (__bf16*)carve(nX * 2);
  __bf16* WqT  = (__bf16*)carve(nW3 * 2);
  __bf16* WkT  = (__bf16*)carve(nW3 * 2);
  __bf16* WvT  = (__bf16*)carve(nW3 * 2);
  __bf16* WpT  = (__bf16*)carve(nW1 * 2);
  __bf16* qbuf = (__bf16*)carve(nX * 2);          // [M,B,H,N,hd]
  __bf16* kbuf = (__bf16*)carve(nX * 2);          // [M,B,H,N,hd]
  __bf16* vtbf = (__bf16*)carve(nX * 2);          // [M,B,H,hd,N]
  __bf16* fub  = (__bf16*)carve(nF * 2);          // [B,N,C]

  // 1) convert / transpose
  cvt_bf16_kernel<<<(int)((nX + 255) / 256), 256, 0, stream>>>(x, xb, (int)nX);
  transpose_bf16_kernel<<<(int)((nW3 + 255) / 256), 256, 0, stream>>>(Wq, WqT, cM);
  transpose_bf16_kernel<<<(int)((nW3 + 255) / 256), 256, 0, stream>>>(Wk, WkT, cM);
  transpose_bf16_kernel<<<(int)((nW3 + 255) / 256), 256, 0, stream>>>(Wv, WvT, cM);
  transpose_bf16_kernel<<<(int)((nW1 + 255) / 256), 256, 0, stream>>>(Wp, WpT, 1);

  // 2) QKV projections (V stored hd-major for P*V B-fragments)
  dim3 gProj(BN / 16, cC / 16, cM);
  qkv_proj_kernel<<<gProj, 32, 0, stream>>>(xb, WqT, bq, qbuf, 0);
  qkv_proj_kernel<<<gProj, 32, 0, stream>>>(xb, WkT, bk, kbuf, 0);
  qkv_proj_kernel<<<gProj, 32, 0, stream>>>(xb, WvT, bv, vtbf, 1);

  // 3) 9-pair attention + fuse
  dim3 gAttn(cN / 16, cB * cH);
  cross_attn_kernel<<<gAttn, 32, 0, stream>>>(qbuf, kbuf, vtbf, mw, fub);

  // 4) output projection
  dim3 gOut(BN / 16, cC / 16);
  out_proj_kernel<<<gOut, 32, 0, stream>>>(fub, WpT, bp, out);
}
